// Transformer_10093173146253
// MI455X (gfx1250) — compile-verified, matrix operand order checked by reading, not run on hardware
//
#include <hip/hip_runtime.h>
#include <hip/hip_bf16.h>
#include <math.h>

typedef __bf16 bf16_t;
typedef __attribute__((ext_vector_type(2)))  __bf16 v2bf;
typedef __attribute__((ext_vector_type(16))) __bf16 v16bf;
typedef __attribute__((ext_vector_type(8)))  float  v8f;

#define DIMD  512
#define SEQ   1024
#define NBATCH 8
#define NHEADS 8
#define HDIM  64

// native fptrunc (RNE) -> lets LLVM pick v_cvt_pk_bf16_f32 when available
__device__ __forceinline__ unsigned short f2bf(float f) {
    __bf16 h = (__bf16)f;
    return __builtin_bit_cast(unsigned short, h);
}
__device__ __forceinline__ unsigned int f2bf2(float a, float b) {
    v2bf t;
    t[0] = (__bf16)a;
    t[1] = (__bf16)b;
    return __builtin_bit_cast(unsigned int, t);
}

union FragAB { v16bf v; uint4 u[2]; };
union U16x8  { uint4 u; unsigned short s[8]; };

__device__ __forceinline__ v8f wmma_bf16(const FragAB& a, const FragAB& b, v8f c) {
    // D = A(16x32 bf16) * B(32x16 bf16) + C(16x16 f32)
    return __builtin_amdgcn_wmma_f32_16x16x32_bf16(false, a.v, false, b.v,
                                                   (short)0, c, false, false);
}

// ---------------------------------------------------------------------------
// GroupNorm: q [8,512,32,32] -> token-major out [8192,512] (fused transpose)
// one block per (batch, group); 32 groups of 16 channels
// ---------------------------------------------------------------------------
__global__ __launch_bounds__(256) void gn_k(const float* __restrict__ Q,
                                            const float* __restrict__ w,
                                            const float* __restrict__ bgn,
                                            float* __restrict__ Out) {
    __shared__ float s1[256], s2[256];
    const int b = blockIdx.x >> 5;   // 0..7
    const int g = blockIdx.x & 31;   // 0..31
    const float* base = Q + ((size_t)b * DIMD + (size_t)g * 16) * 1024;
    float sum = 0.f, sq = 0.f;
    for (int e = threadIdx.x; e < 16 * 1024; e += 256) {
        float v = base[e];
        sum += v; sq += v * v;
    }
    s1[threadIdx.x] = sum; s2[threadIdx.x] = sq;
    __syncthreads();
    for (int s = 128; s > 0; s >>= 1) {
        if (threadIdx.x < s) { s1[threadIdx.x] += s1[threadIdx.x + s];
                               s2[threadIdx.x] += s2[threadIdx.x + s]; }
        __syncthreads();
    }
    const float mu   = s1[0] * (1.0f / 16384.0f);
    const float var  = s2[0] * (1.0f / 16384.0f) - mu * mu;
    const float rstd = rsqrtf(var + 1e-6f);
    for (int e = threadIdx.x; e < 16 * 1024; e += 256) {
        int c   = g * 16 + (e >> 10);
        int pix = e & 1023;
        float v = (base[e] - mu) * rstd;
        Out[((size_t)b * 1024 + pix) * DIMD + c] = v * w[c] + bgn[c];
    }
}

// ---------------------------------------------------------------------------
// LayerNorm over last dim (512); one block per token
// ---------------------------------------------------------------------------
__global__ __launch_bounds__(256) void ln_k(const float* __restrict__ X,
                                            const float* __restrict__ w,
                                            const float* __restrict__ bg,
                                            float* __restrict__ Out) {
    __shared__ float s1[256], s2[256];
    const size_t t = blockIdx.x;
    const float* xp = X + t * DIMD;
    float v0 = xp[threadIdx.x], v1 = xp[threadIdx.x + 256];
    s1[threadIdx.x] = v0 + v1;
    s2[threadIdx.x] = v0 * v0 + v1 * v1;
    __syncthreads();
    for (int s = 128; s > 0; s >>= 1) {
        if (threadIdx.x < s) { s1[threadIdx.x] += s1[threadIdx.x + s];
                               s2[threadIdx.x] += s2[threadIdx.x + s]; }
        __syncthreads();
    }
    const float mu   = s1[0] * (1.0f / DIMD);
    const float var  = s2[0] * (1.0f / DIMD) - mu * mu;
    const float rstd = rsqrtf(var + 1e-5f);
    float* op = Out + t * DIMD;
    op[threadIdx.x]       = (v0 - mu) * rstd * w[threadIdx.x]       + bg[threadIdx.x];
    op[threadIdx.x + 256] = (v1 - mu) * rstd * w[threadIdx.x + 256] + bg[threadIdx.x + 256];
}

// ---------------------------------------------------------------------------
// WMMA GEMM: C[M,N] = A[M,K] * W[N,K]^T (+bias) (+epilogue)
// 128x64 macro-tile per block, 256 threads = 8 wave32; wave computes 32x32
// via 4 WMMAs per k-step (2 A-frags x 2 B-frags).
// EPI: 0 = f32 out, 1 = f32 out + res, 2 = bf16 out, 3 = NCHW transpose + res
// ---------------------------------------------------------------------------
template <int EPI>
__global__ __launch_bounds__(256) void gemm_wmma_k(
        const float* __restrict__ A, const float* __restrict__ W,
        const float* __restrict__ bias, const float* __restrict__ res,
        float* __restrict__ outF, unsigned short* __restrict__ outB,
        int M, int N, int K) {
    __shared__ __align__(16) unsigned short As[128][40];   // 80B row stride
    __shared__ __align__(16) unsigned short Bs[64][40];
    const int tid  = threadIdx.x;
    const int lane = tid & 31, wave = tid >> 5;
    const int half = lane >> 4, l16 = lane & 15;
    const int wm = wave & 3, wn = wave >> 2;
    const int m0 = blockIdx.x * 128, n0 = blockIdx.y * 64;
    const int lrow = tid >> 3;           // 0..31
    const int lcol = (tid & 7) * 4;      // 0..28
    v8f acc00, acc01, acc10, acc11;
    { v8f z = {}; acc00 = z; acc01 = z; acc10 = z; acc11 = z; }

    for (int k0 = 0; k0 < K; k0 += 32) {
#pragma unroll
        for (int rr = lrow; rr < 128; rr += 32) {
            int gr = m0 + rr;
            float4 fa = make_float4(0.f, 0.f, 0.f, 0.f);
            if (gr < M) fa = *(const float4*)(A + (size_t)gr * K + k0 + lcol);
            *(unsigned int*)&As[rr][lcol]     = f2bf2(fa.x, fa.y);
            *(unsigned int*)&As[rr][lcol + 2] = f2bf2(fa.z, fa.w);
        }
#pragma unroll
        for (int rr = lrow; rr < 64; rr += 32) {
            float4 fw = *(const float4*)(W + (size_t)(n0 + rr) * K + k0 + lcol);
            *(unsigned int*)&Bs[rr][lcol]     = f2bf2(fw.x, fw.y);
            *(unsigned int*)&Bs[rr][lcol + 2] = f2bf2(fw.z, fw.w);
        }
        if (k0 + 32 < K) {   // L2 prefetch of next weight tile -> global_prefetch_b8
            __builtin_prefetch(W + (size_t)(n0 + lrow) * K + k0 + 32 + lcol, 0, 1);
        }
        __syncthreads();
        FragAB a0, a1, b0, b1;
        const int ar = wm * 32 + l16;
        a0.u[0] = *(const uint4*)&As[ar][8 * half];
        a0.u[1] = *(const uint4*)&As[ar][16 + 8 * half];
        a1.u[0] = *(const uint4*)&As[ar + 16][8 * half];
        a1.u[1] = *(const uint4*)&As[ar + 16][16 + 8 * half];
        const int br = wn * 32 + l16;
        b0.u[0] = *(const uint4*)&Bs[br][8 * half];
        b0.u[1] = *(const uint4*)&Bs[br][16 + 8 * half];
        b1.u[0] = *(const uint4*)&Bs[br + 16][8 * half];
        b1.u[1] = *(const uint4*)&Bs[br + 16][16 + 8 * half];
        acc00 = wmma_bf16(a0, b0, acc00);
        acc01 = wmma_bf16(a0, b1, acc01);
        acc10 = wmma_bf16(a1, b0, acc10);
        acc11 = wmma_bf16(a1, b1, acc11);
        __syncthreads();
    }

    const int nc0 = n0 + wn * 32 + l16;
    const int nc1 = nc0 + 16;
    const float bv0 = bias ? bias[nc0] : 0.f;
    const float bv1 = bias ? bias[nc1] : 0.f;
#pragma unroll
    for (int mi = 0; mi < 2; ++mi) {
        v8f& c0 = mi ? acc10 : acc00;
        v8f& c1 = mi ? acc11 : acc01;
#pragma unroll
        for (int r = 0; r < 8; ++r) {
            int row = m0 + wm * 32 + mi * 16 + r + 8 * half;
            if (row >= M) continue;
            float v0 = c0[r] + bv0;
            float v1 = c1[r] + bv1;
            if (EPI == 0) {
                outF[(size_t)row * N + nc0] = v0;
                outF[(size_t)row * N + nc1] = v1;
            } else if (EPI == 1) {
                outF[(size_t)row * N + nc0] = v0 + res[(size_t)row * N + nc0];
                outF[(size_t)row * N + nc1] = v1 + res[(size_t)row * N + nc1];
            } else if (EPI == 2) {
                outB[(size_t)row * N + nc0] = f2bf(v0);
                outB[(size_t)row * N + nc1] = f2bf(v1);
            } else {  // EPI == 3: token-major -> NCHW, + residual q
                int bb = row >> 10, pix = row & 1023;
                size_t o0 = ((size_t)(bb * 512 + nc0)) * 1024 + pix;
                size_t o1 = ((size_t)(bb * 512 + nc1)) * 1024 + pix;
                outF[o0] = v0 + res[o0];
                outF[o1] = v1 + res[o1];
            }
        }
    }
}

// ---------------------------------------------------------------------------
// fc0 + GEGLU fused: out[:,n] = (A*W[n]+b[n]) * gelu(A*W[n+2048]+b[n+2048])
// 128x64 macro-tile; wave computes 32x32 of both halves (8 WMMAs / k-step).
// ---------------------------------------------------------------------------
__global__ __launch_bounds__(256) void fc0_geglu_k(
        const float* __restrict__ A, const float* __restrict__ W,
        const float* __restrict__ bias, float* __restrict__ OutG,
        int M, int K, int NH) {
    __shared__ __align__(16) unsigned short As[128][40];
    __shared__ __align__(16) unsigned short Bs[64][40];
    __shared__ __align__(16) unsigned short Cs[64][40];
    const int tid  = threadIdx.x;
    const int lane = tid & 31, wave = tid >> 5;
    const int half = lane >> 4, l16 = lane & 15;
    const int wm = wave & 3, wn = wave >> 2;
    const int m0 = blockIdx.x * 128, n0 = blockIdx.y * 64;
    const int lrow = tid >> 3;
    const int lcol = (tid & 7) * 4;
    v8f x00, x01, x10, x11, h00, h01, h10, h11;
    { v8f z = {}; x00 = z; x01 = z; x10 = z; x11 = z;
                  h00 = z; h01 = z; h10 = z; h11 = z; }

    for (int k0 = 0; k0 < K; k0 += 32) {
#pragma unroll
        for (int rr = lrow; rr < 128; rr += 32) {
            int gr = m0 + rr;
            float4 fa = make_float4(0.f, 0.f, 0.f, 0.f);
            if (gr < M) fa = *(const float4*)(A + (size_t)gr * K + k0 + lcol);
            *(unsigned int*)&As[rr][lcol]     = f2bf2(fa.x, fa.y);
            *(unsigned int*)&As[rr][lcol + 2] = f2bf2(fa.z, fa.w);
        }
#pragma unroll
        for (int rr = lrow; rr < 64; rr += 32) {
            float4 fw = *(const float4*)(W + (size_t)(n0 + rr) * K + k0 + lcol);
            *(unsigned int*)&Bs[rr][lcol]     = f2bf2(fw.x, fw.y);
            *(unsigned int*)&Bs[rr][lcol + 2] = f2bf2(fw.z, fw.w);
            float4 fg = *(const float4*)(W + (size_t)(NH + n0 + rr) * K + k0 + lcol);
            *(unsigned int*)&Cs[rr][lcol]     = f2bf2(fg.x, fg.y);
            *(unsigned int*)&Cs[rr][lcol + 2] = f2bf2(fg.z, fg.w);
        }
        __syncthreads();
        FragAB a0, a1, b0, b1, c0, c1;
        const int ar = wm * 32 + l16;
        a0.u[0] = *(const uint4*)&As[ar][8 * half];
        a0.u[1] = *(const uint4*)&As[ar][16 + 8 * half];
        a1.u[0] = *(const uint4*)&As[ar + 16][8 * half];
        a1.u[1] = *(const uint4*)&As[ar + 16][16 + 8 * half];
        const int br = wn * 32 + l16;
        b0.u[0] = *(const uint4*)&Bs[br][8 * half];
        b0.u[1] = *(const uint4*)&Bs[br][16 + 8 * half];
        b1.u[0] = *(const uint4*)&Bs[br + 16][8 * half];
        b1.u[1] = *(const uint4*)&Bs[br + 16][16 + 8 * half];
        c0.u[0] = *(const uint4*)&Cs[br][8 * half];
        c0.u[1] = *(const uint4*)&Cs[br][16 + 8 * half];
        c1.u[0] = *(const uint4*)&Cs[br + 16][8 * half];
        c1.u[1] = *(const uint4*)&Cs[br + 16][16 + 8 * half];
        x00 = wmma_bf16(a0, b0, x00);
        x01 = wmma_bf16(a0, b1, x01);
        x10 = wmma_bf16(a1, b0, x10);
        x11 = wmma_bf16(a1, b1, x11);
        h00 = wmma_bf16(a0, c0, h00);
        h01 = wmma_bf16(a0, c1, h01);
        h10 = wmma_bf16(a1, c0, h10);
        h11 = wmma_bf16(a1, c1, h11);
        __syncthreads();
    }

    const int nc0 = n0 + wn * 32 + l16;
    const int nc1 = nc0 + 16;
    const float bx0 = bias[nc0],      bx1 = bias[nc1];
    const float bh0 = bias[NH + nc0], bh1 = bias[NH + nc1];
#pragma unroll
    for (int mi = 0; mi < 2; ++mi) {
        v8f& xa = mi ? x10 : x00;
        v8f& xb = mi ? x11 : x01;
        v8f& ha = mi ? h10 : h00;
        v8f& hb = mi ? h11 : h01;
#pragma unroll
        for (int r = 0; r < 8; ++r) {
            int row = m0 + wm * 32 + mi * 16 + r + 8 * half;
            if (row >= M) continue;
            float a0 = xa[r] + bx0;
            float a1 = xb[r] + bx1;
            float g0 = ha[r] + bh0;
            float g1 = hb[r] + bh1;
            g0 = 0.5f * g0 * (1.0f + erff(g0 * 0.70710678118654752f));
            g1 = 0.5f * g1 * (1.0f + erff(g1 * 0.70710678118654752f));
            OutG[(size_t)row * NH + nc0] = a0 * g0;
            OutG[(size_t)row * NH + nc1] = a1 * g1;
        }
    }
}

// ---------------------------------------------------------------------------
// Flash-style attention (bf16 q/k/v, f32 out). Block: 128 q-rows, 8 waves.
// grid = (Lq/128, heads, batch). Q/K/V token-major [rows,512], head at h*64.
// ---------------------------------------------------------------------------
__global__ __launch_bounds__(256) void attn_k(
        const unsigned short* __restrict__ Qb,
        const unsigned short* __restrict__ Kb,
        const unsigned short* __restrict__ Vb,
        float* __restrict__ Out, int Lk) {
    __shared__ __align__(16) unsigned short Ks[64][72];   // [key][dh]
    __shared__ __align__(16) unsigned short Vt[64][72];   // [dh][key]
    __shared__ __align__(16) unsigned short Ps[128][72];  // [qrow][key]
    const int tid  = threadIdx.x;
    const int lane = tid & 31, wave = tid >> 5;
    const int half = lane >> 4, l16 = lane & 15;
    const int h = blockIdx.y, b = blockIdx.z;
    const int hoff  = h * HDIM;
    const size_t qrow0 = (size_t)b * SEQ + (size_t)blockIdx.x * 128;
    const size_t krow0 = (size_t)b * (size_t)Lk;

    // Q fragments (A layout): rows wave*16 + l16, dh split into 2 k-steps
    FragAB qa[2];
    {
        const unsigned short* qp = Qb + (qrow0 + wave * 16 + l16) * DIMD + hoff;
        qa[0].u[0] = *(const uint4*)(qp + 8 * half);
        qa[0].u[1] = *(const uint4*)(qp + 16 + 8 * half);
        qa[1].u[0] = *(const uint4*)(qp + 32 + 8 * half);
        qa[1].u[1] = *(const uint4*)(qp + 48 + 8 * half);
    }
    float mi[8], li[8];
#pragma unroll
    for (int r = 0; r < 8; ++r) { mi[r] = -1e30f; li[r] = 0.f; }
    v8f O[4];
    { v8f z = {}; O[0] = z; O[1] = z; O[2] = z; O[3] = z; }

    for (int j0 = 0; j0 < Lk; j0 += 64) {
        {   // load K tile and transposed V tile
            int key = tid >> 2;
            int dc  = (tid & 3) * 16;
            U16x8 k0v, k1v, v0v, v1v;
            int gk = j0 + key;
            if (gk < Lk) {
                const unsigned short* kp = Kb + (krow0 + gk) * DIMD + hoff + dc;
                const unsigned short* vp = Vb + (krow0 + gk) * DIMD + hoff + dc;
                k0v.u = *(const uint4*)(kp);  k1v.u = *(const uint4*)(kp + 8);
                v0v.u = *(const uint4*)(vp);  v1v.u = *(const uint4*)(vp + 8);
            } else {
                k0v.u = make_uint4(0, 0, 0, 0); k1v.u = k0v.u;
                v0v.u = k0v.u; v1v.u = k0v.u;
            }
            *(uint4*)&Ks[key][dc]     = k0v.u;
            *(uint4*)&Ks[key][dc + 8] = k1v.u;
#pragma unroll
            for (int i = 0; i < 8; ++i) {
                Vt[dc + i][key]     = v0v.s[i];
                Vt[dc + 8 + i][key] = v1v.s[i];
            }
        }
        __syncthreads();

        // S = Q * K^T  (wave: 16 rows x 64 cols)
        v8f S[4];
        { v8f z = {}; S[0] = z; S[1] = z; S[2] = z; S[3] = z; }
#pragma unroll
        for (int kk = 0; kk < 2; ++kk) {
#pragma unroll
            for (int nt = 0; nt < 4; ++nt) {
                FragAB bf;
                const unsigned short* kp = &Ks[nt * 16 + l16][kk * 32 + 8 * half];
                bf.u[0] = *(const uint4*)kp;
                bf.u[1] = *(const uint4*)(kp + 16);
                S[nt] = wmma_bf16(qa[kk], bf, S[nt]);
            }
        }

        // online softmax (row stats replicated across 16-lane groups)
        float p[4][8];
#pragma unroll
        for (int r = 0; r < 8; ++r) {
            float mx = mi[r];
#pragma unroll
            for (int nt = 0; nt < 4; ++nt) {
                float s = S[nt][r] * 0.125f;
                if (j0 + nt * 16 + l16 >= Lk) s = -1e30f;
                p[nt][r] = s;
                mx = fmaxf(mx, s);
            }
#pragma unroll
            for (int off = 8; off >= 1; off >>= 1)
                mx = fmaxf(mx, __shfl_xor(mx, off, 32));
            float corr = __expf(mi[r] - mx);
            float rs = 0.f;
#pragma unroll
            for (int nt = 0; nt < 4; ++nt) {
                float e = __expf(p[nt][r] - mx);
                p[nt][r] = e;
                rs += e;
            }
#pragma unroll
            for (int off = 8; off >= 1; off >>= 1)
                rs += __shfl_xor(rs, off, 32);
            li[r] = li[r] * corr + rs;
            mi[r] = mx;
#pragma unroll
            for (int dt = 0; dt < 4; ++dt) O[dt][r] *= corr;
        }

        // relayout P: C-layout -> LDS -> A-layout
#pragma unroll
        for (int nt = 0; nt < 4; ++nt)
#pragma unroll
            for (int r = 0; r < 8; ++r)
                Ps[wave * 16 + r + 8 * half][nt * 16 + l16] = f2bf(p[nt][r]);
        __syncthreads();

        // O += P * V
#pragma unroll
        for (int kk = 0; kk < 2; ++kk) {
            FragAB pa;
            const unsigned short* pp = &Ps[wave * 16 + l16][kk * 32 + 8 * half];
            pa.u[0] = *(const uint4*)pp;
            pa.u[1] = *(const uint4*)(pp + 16);
#pragma unroll
            for (int dt = 0; dt < 4; ++dt) {
                FragAB vf;
                const unsigned short* vp = &Vt[dt * 16 + l16][kk * 32 + 8 * half];
                vf.u[0] = *(const uint4*)vp;
                vf.u[1] = *(const uint4*)(vp + 16);
                O[dt] = wmma_bf16(pa, vf, O[dt]);
            }
        }
        __syncthreads();
    }

#pragma unroll
    for (int r = 0; r < 8; ++r) {
        float inv = 1.0f / li[r];
        float* op = Out + (qrow0 + wave * 16 + r + 8 * half) * DIMD + hoff;
#pragma unroll
        for (int dt = 0; dt < 4; ++dt)
            op[dt * 16 + l16] = O[dt][r] * inv;
    }
}

// ---------------------------------------------------------------------------
extern "C" void kernel_launch(void* const* d_in, const int* in_sizes, int n_in,
                              void* d_out, int out_size, void* d_ws, size_t ws_size,
                              hipStream_t stream) {
    const float* q_in      = (const float*)d_in[0];
    const float* kv_in     = (const float*)d_in[1];
    const float* gn_w      = (const float*)d_in[2];
    const float* gn_b      = (const float*)d_in[3];
    const float* conv_in_w = (const float*)d_in[4];
    const float* conv_in_b = (const float*)d_in[5];
    const float* ln0_w     = (const float*)d_in[6];
    const float* ln0_b     = (const float*)d_in[7];
    const float* a1_q      = (const float*)d_in[8];
    const float* a1_k      = (const float*)d_in[9];
    const float* a1_v      = (const float*)d_in[10];
    const float* a1_o      = (const float*)d_in[11];
    const float* a1_ob     = (const float*)d_in[12];
    const float* ln1_w     = (const float*)d_in[13];
    const float* ln1_b     = (const float*)d_in[14];
    const float* a2_q      = (const float*)d_in[15];
    const float* a2_k      = (const float*)d_in[16];
    const float* a2_v      = (const float*)d_in[17];
    const float* a2_o      = (const float*)d_in[18];
    const float* a2_ob     = (const float*)d_in[19];
    const float* ln2_w     = (const float*)d_in[20];
    const float* ln2_b     = (const float*)d_in[21];
    const float* fc0_w     = (const float*)d_in[22];
    const float* fc0_b     = (const float*)d_in[23];
    const float* fc1_w     = (const float*)d_in[24];
    const float* fc1_b     = (const float*)d_in[25];
    const float* conv_out_w= (const float*)d_in[26];
    const float* conv_out_b= (const float*)d_in[27];
    float* out = (float*)d_out;

    const int M = NBATCH * SEQ;          // 8192 tokens
    const int Mkv = NBATCH * 77;         // 616 kv tokens

    char* ws = (char*)d_ws;
    size_t off = 0;
    auto alloc = [&](size_t bytes) {
        void* p = ws + off;
        off += (bytes + 255) & ~(size_t)255;
        return p;
    };
    float* gn_t = (float*)alloc((size_t)M * DIMD * 4);
    float* xbuf = (float*)alloc((size_t)M * DIMD * 4);
    float* nbuf = (float*)alloc((size_t)M * DIMD * 4);
    float* tmp  = (float*)alloc((size_t)M * DIMD * 4);
    unsigned short* qb  = (unsigned short*)alloc((size_t)M * DIMD * 2);
    unsigned short* kb  = (unsigned short*)alloc((size_t)M * DIMD * 2);
    unsigned short* vb  = (unsigned short*)alloc((size_t)M * DIMD * 2);
    unsigned short* kb2 = (unsigned short*)alloc((size_t)Mkv * DIMD * 2);
    unsigned short* vb2 = (unsigned short*)alloc((size_t)Mkv * DIMD * 2);
    float* gbuf = (float*)alloc((size_t)M * 2048 * 4);
    (void)ws_size; (void)in_sizes; (void)n_in; (void)out_size;

    const dim3 blk(256);
    const dim3 gemmGrid(M / 128, DIMD / 64);         // 64 x 8
    const dim3 kvGrid((Mkv + 127) / 128, DIMD / 64); // 5 x 8
    const dim3 attnGrid(SEQ / 128, NHEADS, NBATCH);

    // 1. GroupNorm + NCHW->token-major transpose
    gn_k<<<dim3(NBATCH * 32), blk, 0, stream>>>(q_in, gn_w, gn_b, gn_t);
    // 2. conv_in (1x1) as GEMM
    gemm_wmma_k<0><<<gemmGrid, blk, 0, stream>>>(gn_t, conv_in_w, conv_in_b,
                                                 nullptr, xbuf, nullptr, M, DIMD, DIMD);
    // 3. LN0
    ln_k<<<dim3(M), blk, 0, stream>>>(xbuf, ln0_w, ln0_b, nbuf);
    // 4. self-attn projections -> bf16
    gemm_wmma_k<2><<<gemmGrid, blk, 0, stream>>>(nbuf, a1_q, nullptr, nullptr,
                                                 nullptr, qb, M, DIMD, DIMD);
    gemm_wmma_k<2><<<gemmGrid, blk, 0, stream>>>(nbuf, a1_k, nullptr, nullptr,
                                                 nullptr, kb, M, DIMD, DIMD);
    gemm_wmma_k<2><<<gemmGrid, blk, 0, stream>>>(nbuf, a1_v, nullptr, nullptr,
                                                 nullptr, vb, M, DIMD, DIMD);
    // 5. self-attention
    attn_k<<<attnGrid, blk, 0, stream>>>(qb, kb, vb, tmp, SEQ);
    // 6. out-proj + residual
    gemm_wmma_k<1><<<gemmGrid, blk, 0, stream>>>(tmp, a1_o, a1_ob, xbuf,
                                                 xbuf, nullptr, M, DIMD, DIMD);
    // 7. LN1
    ln_k<<<dim3(M), blk, 0, stream>>>(xbuf, ln1_w, ln1_b, nbuf);
    // 8. cross-attn projections
    gemm_wmma_k<2><<<gemmGrid, blk, 0, stream>>>(nbuf, a2_q, nullptr, nullptr,
                                                 nullptr, qb, M, DIMD, DIMD);
    gemm_wmma_k<2><<<kvGrid, blk, 0, stream>>>(kv_in, a2_k, nullptr, nullptr,
                                               nullptr, kb2, Mkv, DIMD, DIMD);
    gemm_wmma_k<2><<<kvGrid, blk, 0, stream>>>(kv_in, a2_v, nullptr, nullptr,
                                               nullptr, vb2, Mkv, DIMD, DIMD);
    // 9. cross-attention (Lk = 77, masked)
    attn_k<<<attnGrid, blk, 0, stream>>>(qb, kb2, vb2, tmp, 77);
    // 10. out-proj + residual  (xbuf becomes res2)
    gemm_wmma_k<1><<<gemmGrid, blk, 0, stream>>>(tmp, a2_o, a2_ob, xbuf,
                                                 xbuf, nullptr, M, DIMD, DIMD);
    // 11. LN2
    ln_k<<<dim3(M), blk, 0, stream>>>(xbuf, ln2_w, ln2_b, nbuf);
    // 12. fc0 + GEGLU fused
    fc0_geglu_k<<<dim3(M / 128, 2048 / 64), blk, 0, stream>>>(nbuf, fc0_w, fc0_b,
                                                              gbuf, M, DIMD, 2048);
    // 13. fc1 + residual(res2)
    gemm_wmma_k<1><<<gemmGrid, blk, 0, stream>>>(gbuf, fc1_w, fc1_b, xbuf,
                                                 tmp, nullptr, M, DIMD, 2048);
    // 14. conv_out (1x1) + transpose back to NCHW + residual q
    gemm_wmma_k<3><<<gemmGrid, blk, 0, stream>>>(tmp, conv_out_w, conv_out_b, q_in,
                                                 out, nullptr, M, DIMD, DIMD);
}
